// dvsFeature_47004122087992
// MI455X (gfx1250) — matrix operand order repaired and checked by reading, not compiled, not run on hardware
//
#include <hip/hip_runtime.h>
#include <hip/hip_fp16.h>

typedef __attribute__((ext_vector_type(16))) _Float16 v16h;
typedef __attribute__((ext_vector_type(8)))  float    v8f;

#define T_STEPS 128
#define N_BATCH 16

// Loihi CUBA neuron: iDecay=1024, vDecay=128 (of 4096), theta=100
__device__ __forceinline__ float loihi_step(float& u, float& v, float x) {
  u = 0.75f    * u + x;   // (4096-1024)/4096
  v = 0.96875f * v + u;   // (4096-128)/4096
  float s = (v >= 100.0f) ? 1.0f : 0.0f;
  v = v * (1.0f - s);     // reset on spike
  return s;
}

// ---------------------------------------------------------------------------
// 3x3 conv (pad 1) fused with Loihi scan over T.
// DELAY: read input at t-1 (delay_shift applied to previous layer's spikes).
// NT_LAYOUT: write fp16 spikes to [n][t][c] (GEMM A layout); else u8 [n][c][h][w][t].
// ---------------------------------------------------------------------------
template <int CIN, int COUT, int HW, typename InT, typename OutT, bool DELAY, bool NT_LAYOUT>
__global__ void conv3x3_scan(const InT* __restrict__ in, const float* __restrict__ w,
                             float wscale, OutT* __restrict__ out) {
  __shared__ float ws[COUT * CIN * 9];
  for (int i = threadIdx.x; i < COUT * CIN * 9; i += blockDim.x)
    ws[i] = w[i] * wscale;
  __syncthreads();

  const int gid = blockIdx.x * blockDim.x + threadIdx.x;   // exact grid, no bounds check
  const int x  = gid % HW;
  const int y  = (gid / HW) % HW;
  const int oc = (gid / (HW * HW)) % COUT;
  const int n  = gid / (HW * HW * COUT);

  const float* wk = &ws[oc * CIN * 9];

  float u = 0.f, v = 0.f;
  for (int t = 0; t < T_STEPS; ++t) {
    const int tin = DELAY ? (t - 1) : t;
    float acc = 0.f;
    if (tin >= 0) {
#pragma unroll
      for (int ic = 0; ic < CIN; ++ic) {
        const InT* inc = in + (size_t)(n * CIN + ic) * (HW * HW) * T_STEPS;
#pragma unroll
        for (int ky = 0; ky < 3; ++ky) {
          const int yy = y + ky - 1;
          if (yy < 0 || yy >= HW) continue;
#pragma unroll
          for (int kx = 0; kx < 3; ++kx) {
            const int xx = x + kx - 1;
            if (xx < 0 || xx >= HW) continue;
            acc += wk[(ic * 3 + ky) * 3 + kx] *
                   (float)inc[(size_t)(yy * HW + xx) * T_STEPS + tin];
          }
        }
      }
    }
    const float s = loihi_step(u, v, acc);
    if (NT_LAYOUT) {
      out[((size_t)n * T_STEPS + t) * (COUT * HW * HW) + (oc * HW + y) * HW + x] = (OutT)s;
    } else {
      out[(((size_t)(n * COUT + oc) * HW + y) * HW + x) * T_STEPS + t] = (OutT)s;
    }
  }
}

// ---------------------------------------------------------------------------
// 2x2 sum-pool (x pool weight) fused with Loihi scan; input delayed by 1.
// ---------------------------------------------------------------------------
template <int C, int HO>
__global__ void pool2x2_scan(const unsigned char* __restrict__ in,
                             const float* __restrict__ pw,
                             unsigned char* __restrict__ out) {
  const int gid = blockIdx.x * blockDim.x + threadIdx.x;
  const int x = gid % HO;
  const int y = (gid / HO) % HO;
  const int c = (gid / (HO * HO)) % C;
  const int n = gid / (HO * HO * C);
  const int HI = 2 * HO;

  const float scale = *pw;
  const unsigned char* p =
      in + (((size_t)(n * C + c) * HI + 2 * y) * HI + 2 * x) * T_STEPS;
  unsigned char* o = out + (((size_t)(n * C + c) * HO + y) * HO + x) * T_STEPS;

  float u = 0.f, v = 0.f;
  for (int t = 0; t < T_STEPS; ++t) {
    const int tin = t - 1;
    float acc = 0.f;
    if (tin >= 0) {
      float sum = (float)p[tin] + (float)p[T_STEPS + tin] +
                  (float)p[(size_t)HI * T_STEPS + tin] +
                  (float)p[(size_t)HI * T_STEPS + T_STEPS + tin];
      acc = sum * scale;
    }
    o[t] = (unsigned char)loihi_step(u, v, acc);
  }
}

// ---------------------------------------------------------------------------
// fp32 -> fp16 weight conversion for fc1_w
// ---------------------------------------------------------------------------
__global__ void f32_to_f16(const float* __restrict__ src, _Float16* __restrict__ dst, int n) {
  const int i = blockIdx.x * blockDim.x + threadIdx.x;
  if (i < n) dst[i] = (_Float16)src[i];
}

// ---------------------------------------------------------------------------
// FC GEMM via v_wmma_f32_16x16x32_f16.
//   Y[r=(n,t)][o] = sum_c A[r][c] * W[o][c],  A[r][c] = S5h[n][t-1][c] (0 at t=0)
// Each wave computes a 16x64 output tile: one shared A fragment feeds 4
// independent WMMAs per K=32 step (no D->A/B hazards between them).
// Per-wave private LDS slices; no block barriers -- same-wave LDS ops are
// architecturally in-order, explicit s_wait_dscnt covers stage->consume.
// ---------------------------------------------------------------------------
__global__ void fc_gemm_wmma(const _Float16* __restrict__ A,   // [16*128][3200] spikes (undelayed)
                             const _Float16* __restrict__ W,   // [512][3200] fp16 weights
                             float* __restrict__ Y) {          // [2048][512]
  __shared__ __align__(16) _Float16 As[4][16 * 32];   //  4 KB: A tile per wave
  __shared__ __align__(16) _Float16 Bs[4][64 * 32];   // 16 KB: 64x32 B tile per wave

  const int wave = threadIdx.x >> 5;
  const int lane = threadIdx.x & 31;
  const int gw   = blockIdx.x * 4 + wave;      // 1024 waves total
  const int rowTile  = gw >> 3;                // 0..127  (n,t rows)
  const int colGroup = gw & 7;                 // 0..7    (64-wide column group)
  const int rowBase = rowTile * 16;
  const int colBase = colGroup * 64;

  // fragment lane mapping (ISA 7.12.2, 16-bit A 16x32 / B 32x16 / f32 C 16x16)
  const int hi  = lane >> 4;                   // lane half
  const int m   = lane & 15;                   // A row / B col / D col index
  const int akb = hi * 8;                      // half the A K-base (K 2*akb..2*akb+15)

  uint4* AsQ = (uint4*)&As[wave][0];           // 64 uint4 (8 halves each)
  uint4* BsQ = (uint4*)&Bs[wave][0];           // 256 uint4

  // ---- A staging: 64 uint4 chunks; lane covers q = lane, lane+32.
  // chunk q -> row q>>2 (delay_shift: row t uses S5h[n][t-1]; t==0 row is zero,
  // written once below and never overwritten in the loop).
  const _Float16* aSrc[2];
#pragma unroll
  for (int i = 0; i < 2; ++i) {
    const int q   = lane + 32 * i;
    const int row = q >> 2;
    const int r   = rowBase + row;
    const int tt  = r & 127;
    aSrc[i] = (tt == 0) ? nullptr
            : A + ((size_t)(r >> 7) * T_STEPS + (tt - 1)) * 3200 + (q & 3) * 8;
    if (!aSrc[i]) {
      uint4 z = {0u, 0u, 0u, 0u};
      AsQ[q] = z;
    }
  }

  // ---- B staging: 256 uint4 chunks; lane covers q = lane + 32*i (i=0..7).
  // chunk q -> weight row o = q>>2; adjacent lanes touch adjacent 16B chunks.
  const _Float16* bSrc[8];
#pragma unroll
  for (int i = 0; i < 8; ++i) {
    const int q = lane + 32 * i;
    bSrc[i] = W + (size_t)(colBase + (q >> 2)) * 3200 + (q & 3) * 8;
  }

  v8f c0 = {}, c1 = {}, c2 = {}, c3 = {};

  for (int k0 = 0; k0 < 3200; k0 += 32) {
    // stage A (skip permanently-zero rows)
#pragma unroll
    for (int i = 0; i < 2; ++i)
      if (aSrc[i]) AsQ[lane + 32 * i] = *(const uint4*)(aSrc[i] + k0);
    // stage B
#pragma unroll
    for (int i = 0; i < 8; ++i)
      BsQ[lane + 32 * i] = *(const uint4*)(bSrc[i] + k0);

    if (k0 + 32 < 3200) {                      // global_prefetch_b8 of next K slice
      if (aSrc[0]) __builtin_prefetch(aSrc[0] + k0 + 32, 0, 1);
      __builtin_prefetch(bSrc[0] + k0 + 32, 0, 1);
      __builtin_prefetch(bSrc[4] + k0 + 32, 0, 1);
    }

    // same-wave LDS is in-order; wait for staging stores, fence the compiler
    asm volatile("s_wait_dscnt 0x0" ::: "memory");

    // A fragment: elems 0..15 = K 2*akb..2*akb+15 (16*hi..16*hi+15)
    union { v16h h; uint4 q[2]; } ua;
    ua.q[0] = AsQ[m * 4 + (akb >> 2)];
    ua.q[1] = AsQ[m * 4 + 1 + (akb >> 2)];

    // 4 B fragments (sub-tiles s=0..3): lane col n=m, elems = K 16*hi..16*hi+15
#pragma unroll
    for (int s = 0; s < 4; ++s) {
      union { v16h h; uint4 q[2]; } ub;
      const int ob = (s * 16 + m) * 4 + hi * 2;
      ub.q[0] = BsQ[ob];
      ub.q[1] = BsQ[ob + 1];
      v8f& cc = (s == 0) ? c0 : (s == 1) ? c1 : (s == 2) ? c2 : c3;
      cc = __builtin_amdgcn_wmma_f32_16x16x32_f16(false, ua.h, false, ub.h,
                                                  (short)0, cc, false, false);
    }
    asm volatile("" ::: "memory");             // keep next-iter stores behind reads
  }

  // D layout: VGPR j -> row j + 8*hi, col = lane&15
#pragma unroll
  for (int s = 0; s < 4; ++s) {
    const v8f& cc = (s == 0) ? c0 : (s == 1) ? c1 : (s == 2) ? c2 : c3;
#pragma unroll
    for (int j = 0; j < 8; ++j) {
      const int mm = j + hi * 8;
      Y[(size_t)(rowBase + mm) * 512 + colBase + s * 16 + m] = cc[j];
    }
  }
}

// ---------------------------------------------------------------------------
// Final FC Loihi scan + output delay_shift: out[n][o][t] = s[t-1]
// ---------------------------------------------------------------------------
__global__ void fc_scan(const float* __restrict__ Y, float* __restrict__ out) {
  const int gid = blockIdx.x * blockDim.x + threadIdx.x;   // 0..8191
  const int o = gid & 511;
  const int n = gid >> 9;
  const float* yp = Y + (size_t)n * T_STEPS * 512 + o;
  float* op = out + ((size_t)n * 512 + o) * T_STEPS;

  float u = 0.f, v = 0.f, prev = 0.f;
  for (int t = 0; t < T_STEPS; ++t) {
    op[t] = prev;                              // delay_shift on output
    prev = loihi_step(u, v, yp[(size_t)t * 512]);
  }
}

// ---------------------------------------------------------------------------
extern "C" void kernel_launch(void* const* d_in, const int* in_sizes, int n_in,
                              void* d_out, int out_size, void* d_ws, size_t ws_size,
                              hipStream_t stream) {
  (void)in_sizes; (void)n_in; (void)out_size; (void)ws_size;

  const float* spike = (const float*)d_in[0];   // [16,2,40,40,128]
  const float* w1    = (const float*)d_in[1];   // [8,2,3,3]
  const float* w2    = (const float*)d_in[2];   // [16,8,3,3]
  const float* w3    = (const float*)d_in[3];   // [32,16,3,3]
  const float* pw1   = (const float*)d_in[4];   // scalar
  const float* pw2   = (const float*)d_in[5];   // scalar
  const float* wfc   = (const float*)d_in[6];   // [512,3200]
  float* out = (float*)d_out;                   // [16,512,1,1,128]

  char* ws = (char*)d_ws;
  size_t off = 0;
  auto alloc = [&](size_t bytes) {
    size_t o = off;
    off += (bytes + 255) & ~(size_t)255;
    return o;
  };

  unsigned char* S1 = (unsigned char*)(ws + alloc((size_t)16 * 8  * 40 * 40 * 128));
  unsigned char* S2 = (unsigned char*)(ws + alloc((size_t)16 * 8  * 20 * 20 * 128));
  unsigned char* S3 = (unsigned char*)(ws + alloc((size_t)16 * 16 * 20 * 20 * 128));
  unsigned char* S4 = (unsigned char*)(ws + alloc((size_t)16 * 16 * 10 * 10 * 128));
  _Float16*      S5 = (_Float16*)     (ws + alloc((size_t)16 * 128 * 3200 * 2));
  _Float16*      Wh = (_Float16*)     (ws + alloc((size_t)512 * 3200 * 2));
  float*         Yb = (float*)        (ws + alloc((size_t)2048 * 512 * 4));

  // fc weights -> fp16
  f32_to_f16<<<(512 * 3200 + 255) / 256, 256, 0, stream>>>(wfc, Wh, 512 * 3200);

  // conv1 (x20) + scan : [16,2,40,40,T] -> S1 [16,8,40,40,T]
  conv3x3_scan<2, 8, 40, float, unsigned char, false, false>
      <<<800, 256, 0, stream>>>(spike, w1, 20.0f, S1);
  // pool1 + scan (reads t-1) : -> S2 [16,8,20,20,T]
  pool2x2_scan<8, 20><<<200, 256, 0, stream>>>(S1, pw1, S2);
  // conv2 (x100) + scan (reads t-1) : -> S3 [16,16,20,20,T]
  conv3x3_scan<8, 16, 20, unsigned char, unsigned char, true, false>
      <<<400, 256, 0, stream>>>(S2, w2, 100.0f, S3);
  // pool2 + scan (reads t-1) : -> S4 [16,16,10,10,T]
  pool2x2_scan<16, 10><<<100, 256, 0, stream>>>(S3, pw2, S4);
  // conv3 (x100) + scan (reads t-1) : -> S5 fp16 [n][t][c=3200] (GEMM A layout)
  conv3x3_scan<16, 32, 10, unsigned char, _Float16, true, true>
      <<<200, 256, 0, stream>>>(S4, w3, 100.0f, S5);

  // fc1 GEMM (delay handled inside via t-1 row fetch): Y [2048][512]
  fc_gemm_wmma<<<256, 128, 0, stream>>>(S5, Wh, Yb);

  // final scan + output delay shift
  fc_scan<<<32, 256, 0, stream>>>(Yb, out);
}
